// MultiHeadAttention_26431228739713
// MI455X (gfx1250) — compile-verified, matrix operand order checked by reading, not run on hardware
//
#include <hip/hip_runtime.h>
#include <hip/hip_bf16.h>

typedef __attribute__((ext_vector_type(16))) _Float16 v16h;
typedef __attribute__((ext_vector_type(8)))  _Float16 v8h;
typedef __attribute__((ext_vector_type(4)))  _Float16 v4h;
typedef __attribute__((ext_vector_type(8)))  float    v8f;
typedef __attribute__((ext_vector_type(4)))  float    v4f;

#define D_IN   1024
#define D_OUT  1024
#define NHEADS 16
#define HDIM   64
#define SEQ    2048
#define BATCH  2
#define MTOT   (BATCH * SEQ)   // 4096

// ---------------------------------------------------------------------------
// WMMA helper: D = A(16x32 f16) * B(32x16 f16) + C(16x16 f32)
// ---------------------------------------------------------------------------
static __device__ __forceinline__ v8f wmma_f16(v16h a, v16h b, v8f c) {
  return __builtin_amdgcn_wmma_f32_16x16x32_f16(false, a, false, b, (short)0, c,
                                                false, false);
}

// Fragment loader for both A (row-major tile) and B (from pre-transposed
// matrix, so the "column" is a contiguous row). Per CDNA5 ISA 16-bit A layout:
// lane<16: row=lane, K = {8*half+0..7, 16+8*half+0..7} relative to k0.
static __device__ __forceinline__ v16h load_frag(const _Float16* __restrict__ tile,
                                                 int ld, int k0) {
  int lane = threadIdx.x & 31;
  int r = lane & 15;
  int half = lane >> 4;
  const _Float16* p = tile + (size_t)r * ld + k0 + 8 * half;
  v8h lo = *(const v8h*)(p);        // K = k0 + 8*half + 0..7
  v8h hi = *(const v8h*)(p + 16);   // K = k0 + 16 + 8*half + 0..7
  v16h f;
#pragma unroll
  for (int i = 0; i < 8; ++i) { f[i] = lo[i]; f[i + 8] = hi[i]; }
  return f;
}

// 16-lane-group reductions (match C-layout: a row lives in one 16-lane half)
static __device__ __forceinline__ float rmax16(float v) {
#pragma unroll
  for (int m = 1; m < 16; m <<= 1) v = fmaxf(v, __shfl_xor(v, m, 32));
  return v;
}
static __device__ __forceinline__ float rsum16(float v) {
#pragma unroll
  for (int m = 1; m < 16; m <<= 1) v += __shfl_xor(v, m, 32);
  return v;
}

// ---------------------------------------------------------------------------
// Kernel 1: fp32 -> f16 convert (x), vectorized x4
// ---------------------------------------------------------------------------
__global__ void cvt_f32_f16(const float* __restrict__ in,
                            _Float16* __restrict__ out, int n4) {
  int i = blockIdx.x * blockDim.x + threadIdx.x;
  if (i < n4) {
    v4f v = *(const v4f*)(in + (size_t)i * 4);
    v4h h;
    h[0] = (_Float16)v.x; h[1] = (_Float16)v.y;
    h[2] = (_Float16)v.z; h[3] = (_Float16)v.w;
    *(v4h*)(out + (size_t)i * 4) = h;
  }
}

// ---------------------------------------------------------------------------
// Kernel 2: weight transpose + convert: W[k][n] f32 -> WT[n][k] f16 (1024x1024)
// ---------------------------------------------------------------------------
__global__ void transpose_cvt_w(const float* __restrict__ W,
                                _Float16* __restrict__ WT) {
  __shared__ _Float16 tile[32][33];
  int n0 = blockIdx.x * 32, k0 = blockIdx.y * 32;
  int tx = threadIdx.x, ty = threadIdx.y;
#pragma unroll
  for (int i = 0; i < 4; ++i)
    tile[ty * 4 + i][tx] = (_Float16)W[(size_t)(k0 + ty * 4 + i) * 1024 + n0 + tx];
  __syncthreads();
#pragma unroll
  for (int i = 0; i < 4; ++i)
    WT[(size_t)(n0 + ty * 4 + i) * 1024 + k0 + tx] = tile[tx][ty * 4 + i];
}

// ---------------------------------------------------------------------------
// Kernel 3: fused QKV projection GEMM (f16 in, f32 accum, f16 out)
// wave -> 16 rows x 64 cols for each of Q,K,V. Block = 8 waves stacked in M
// sharing identical B addresses (WGP$ reuse).
// ---------------------------------------------------------------------------
__global__ __launch_bounds__(256) void qkv_gemm(
    const _Float16* __restrict__ xh, const _Float16* __restrict__ WqT,
    const _Float16* __restrict__ WkT, const _Float16* __restrict__ WvT,
    _Float16* __restrict__ Q, _Float16* __restrict__ K,
    _Float16* __restrict__ V) {
  int wave = threadIdx.x >> 5, lane = threadIdx.x & 31;
  int nstrip = blockIdx.x & 15;
  int mtile = (blockIdx.x >> 4) * 8 + wave;     // 0..255
  v8f accq[4] = {}, acck[4] = {}, accv[4] = {};
  const _Float16* A = xh + (size_t)mtile * 16 * D_IN;

  for (int k0 = 0; k0 < D_IN; k0 += 32) {
    v16h a = load_frag(A, D_IN, k0);
#pragma unroll
    for (int t = 0; t < 4; ++t) {
      int n0 = nstrip * 64 + t * 16;
      v16h bq = load_frag(WqT + (size_t)n0 * D_IN, D_IN, k0);
      accq[t] = wmma_f16(a, bq, accq[t]);
      v16h bk = load_frag(WkT + (size_t)n0 * D_IN, D_IN, k0);
      acck[t] = wmma_f16(a, bk, acck[t]);
      v16h bv = load_frag(WvT + (size_t)n0 * D_IN, D_IN, k0);
      accv[t] = wmma_f16(a, bv, accv[t]);
    }
  }

  int half = lane >> 4, col = lane & 15;
#pragma unroll
  for (int t = 0; t < 4; ++t) {
    int n = nstrip * 64 + t * 16 + col;
    int h = n >> 6, dh = n & 63;
#pragma unroll
    for (int j = 0; j < 8; ++j) {
      int m = mtile * 16 + j + 8 * half;
      int b = m >> 11, s = m & 2047;
      size_t idx = ((size_t)(b * NHEADS + h) * SEQ + s) * HDIM + dh;
      Q[idx] = (_Float16)accq[t][j];
      K[idx] = (_Float16)acck[t][j];
      V[idx] = (_Float16)accv[t][j];
    }
  }
}

// ---------------------------------------------------------------------------
// Kernel 4: V[bh][s][d] -> VT[bh][d][s] (f16 tiled transpose)
// ---------------------------------------------------------------------------
__global__ void transpose_v(const _Float16* __restrict__ V,
                            _Float16* __restrict__ VT) {
  __shared__ _Float16 tile[32][33];
  int bh = blockIdx.z;
  int d0 = blockIdx.x * 32, s0 = blockIdx.y * 32;
  const _Float16* Vb = V + (size_t)bh * SEQ * HDIM;
  _Float16* Tb = VT + (size_t)bh * SEQ * HDIM;
  int tx = threadIdx.x, ty = threadIdx.y;
#pragma unroll
  for (int i = 0; i < 4; ++i)
    tile[ty * 4 + i][tx] = Vb[(size_t)(s0 + ty * 4 + i) * HDIM + d0 + tx];
  __syncthreads();
#pragma unroll
  for (int i = 0; i < 4; ++i)
    Tb[(size_t)(d0 + ty * 4 + i) * SEQ + s0 + tx] = tile[tx][ty * 4 + i];
}

// ---------------------------------------------------------------------------
// Kernel 5: causal flash attention, one wave = 16 queries of one (b,h).
// QK^T and P.V via WMMA; online softmax in f32; P re-shaped C->A via LDS.
// ---------------------------------------------------------------------------
__global__ __launch_bounds__(256) void flash_attn(
    const _Float16* __restrict__ Q, const _Float16* __restrict__ Kmat,
    const _Float16* __restrict__ VT, _Float16* __restrict__ AO) {
  __shared__ __align__(16) _Float16 pbuf[8][16 * 32];
  int wave = threadIdx.x >> 5, lane = threadIdx.x & 31;
  int wg = blockIdx.x * 8 + wave;
  int qtile = wg & 127;          // SEQ/16
  int bh = wg >> 7;              // 0..31
  int q0 = qtile * 16;
  int half = lane >> 4, col = lane & 15;

  const _Float16* Qh = Q + (size_t)bh * SEQ * HDIM;
  const _Float16* Kh = Kmat + (size_t)bh * SEQ * HDIM;
  const _Float16* Vh = VT + (size_t)bh * HDIM * SEQ;
  _Float16* P = pbuf[wave];

  v16h aq0 = load_frag(Qh + (size_t)q0 * HDIM, HDIM, 0);   // d = 0..31
  v16h aq1 = load_frag(Qh + (size_t)q0 * HDIM, HDIM, 32);  // d = 32..63

  v8f o0 = {}, o1 = {}, o2 = {}, o3 = {};
  float mrow[8], lrow[8];
#pragma unroll
  for (int j = 0; j < 8; ++j) { mrow[j] = -__builtin_inff(); lrow[j] = 0.f; }

  const float sc = 0.03125f;  // 1/sqrt(D_IN) = 1/32 (reference scales by d_in!)

  for (int kb = 0; kb < q0 + 16; kb += 32) {
    // ---- scores: 2 x (16x16) tiles over 32 keys, K-dim = HDIM = 64
    v8f s0 = {}, s1 = {};
    {
      v16h b0 = load_frag(Kh + (size_t)kb * HDIM, HDIM, 0);
      s0 = wmma_f16(aq0, b0, s0);
      v16h b1 = load_frag(Kh + (size_t)kb * HDIM, HDIM, 32);
      s0 = wmma_f16(aq1, b1, s0);
      v16h b2 = load_frag(Kh + (size_t)(kb + 16) * HDIM, HDIM, 0);
      s1 = wmma_f16(aq0, b2, s1);
      v16h b3 = load_frag(Kh + (size_t)(kb + 16) * HDIM, HDIM, 32);
      s1 = wmma_f16(aq1, b3, s1);
    }
    // ---- mask + online softmax, write P (f16) to per-wave LDS tile
#pragma unroll
    for (int j = 0; j < 8; ++j) {
      int q = q0 + j + 8 * half;
      float v0 = s0[j] * sc; if (kb + col > q)      v0 = -__builtin_inff();
      float v1 = s1[j] * sc; if (kb + 16 + col > q) v1 = -__builtin_inff();
      float bm = rmax16(fmaxf(v0, v1));
      float mnew = fmaxf(mrow[j], bm);
      float p0 = __expf(v0 - mnew);
      float p1 = __expf(v1 - mnew);
      float scale = __expf(mrow[j] - mnew);
      mrow[j] = mnew;
      lrow[j] = lrow[j] * scale + rsum16(p0 + p1);
      o0[j] *= scale; o1[j] *= scale; o2[j] *= scale; o3[j] *= scale;
      int m = j + 8 * half;
      P[m * 32 + col]      = (_Float16)p0;
      P[m * 32 + 16 + col] = (_Float16)p1;
    }
    asm volatile("s_wait_dscnt 0x0" ::: "memory");
    // ---- reload P as an A-matrix fragment (16x32)
    v16h ap;
    {
      const _Float16* pp = P + col * 32 + 8 * half;
      v8h lo = *(const v8h*)(pp);
      v8h hi = *(const v8h*)(pp + 16);
#pragma unroll
      for (int i = 0; i < 8; ++i) { ap[i] = lo[i]; ap[i + 8] = hi[i]; }
    }
    // ---- O += P @ V  (B fragments from VT: contiguous along s)
    o0 = wmma_f16(ap, load_frag(Vh + (size_t)0  * SEQ, SEQ, kb), o0);
    o1 = wmma_f16(ap, load_frag(Vh + (size_t)16 * SEQ, SEQ, kb), o1);
    o2 = wmma_f16(ap, load_frag(Vh + (size_t)32 * SEQ, SEQ, kb), o2);
    o3 = wmma_f16(ap, load_frag(Vh + (size_t)48 * SEQ, SEQ, kb), o3);
  }

  // ---- normalize and store AO[b][s][h*64+d] as f16 for the final GEMM
  int b = bh >> 4, h = bh & 15;
#pragma unroll
  for (int j = 0; j < 8; ++j) {
    float inv = 1.0f / lrow[j];
    int s = q0 + j + 8 * half;
    size_t row = ((size_t)(b * SEQ + s)) * D_OUT + h * HDIM;
    AO[row + 0  + col] = (_Float16)(o0[j] * inv);
    AO[row + 16 + col] = (_Float16)(o1[j] * inv);
    AO[row + 32 + col] = (_Float16)(o2[j] * inv);
    AO[row + 48 + col] = (_Float16)(o3[j] * inv);
  }
}

// ---------------------------------------------------------------------------
// Kernel 6: output projection GEMM: out = AO @ Wo + bo (f32 out)
// ---------------------------------------------------------------------------
__global__ __launch_bounds__(256) void out_gemm(
    const _Float16* __restrict__ AO, const _Float16* __restrict__ WoT,
    const float* __restrict__ bo, float* __restrict__ out) {
  int wave = threadIdx.x >> 5, lane = threadIdx.x & 31;
  int nstrip = blockIdx.x & 15;
  int mtile = (blockIdx.x >> 4) * 8 + wave;
  v8f acc[4] = {};
  const _Float16* A = AO + (size_t)mtile * 16 * D_OUT;

  for (int k0 = 0; k0 < D_OUT; k0 += 32) {
    v16h a = load_frag(A, D_OUT, k0);
#pragma unroll
    for (int t = 0; t < 4; ++t) {
      v16h bf = load_frag(WoT + (size_t)(nstrip * 64 + t * 16) * D_OUT, D_OUT, k0);
      acc[t] = wmma_f16(a, bf, acc[t]);
    }
  }
  int half = lane >> 4, col = lane & 15;
#pragma unroll
  for (int t = 0; t < 4; ++t) {
    int n = nstrip * 64 + t * 16 + col;
    float bias = bo[n];
#pragma unroll
    for (int j = 0; j < 8; ++j) {
      int m = mtile * 16 + j + 8 * half;
      out[(size_t)m * D_OUT + n] = acc[t][j] + bias;
    }
  }
}

// ---------------------------------------------------------------------------
extern "C" void kernel_launch(void* const* d_in, const int* in_sizes, int n_in,
                              void* d_out, int out_size, void* d_ws,
                              size_t ws_size, hipStream_t stream) {
  const float* x  = (const float*)d_in[0];
  const float* Wq = (const float*)d_in[1];
  const float* Wk = (const float*)d_in[2];
  const float* Wv = (const float*)d_in[3];
  const float* Wo = (const float*)d_in[4];
  const float* bo = (const float*)d_in[5];
  float* out = (float*)d_out;

  char* ws = (char*)d_ws;
  size_t off = 0;
  _Float16* xh  = (_Float16*)(ws + off); off += (size_t)MTOT * D_IN * 2;   // 8M
  _Float16* WqT = (_Float16*)(ws + off); off += (size_t)D_IN * D_OUT * 2;  // 2M
  _Float16* WkT = (_Float16*)(ws + off); off += (size_t)D_IN * D_OUT * 2;
  _Float16* WvT = (_Float16*)(ws + off); off += (size_t)D_IN * D_OUT * 2;
  _Float16* WoT = (_Float16*)(ws + off); off += (size_t)D_OUT * D_OUT * 2;
  _Float16* Qb  = (_Float16*)(ws + off); off += (size_t)MTOT * D_OUT * 2;  // 8M
  _Float16* Kb  = (_Float16*)(ws + off); off += (size_t)MTOT * D_OUT * 2;
  _Float16* Vb  = (_Float16*)(ws + off); off += (size_t)MTOT * D_OUT * 2;
  _Float16* VTb = (_Float16*)(ws + off); off += (size_t)MTOT * D_OUT * 2;
  _Float16* AOb = (_Float16*)(ws + off); off += (size_t)MTOT * D_OUT * 2;

  dim3 tb(32, 8);

  // Stage 0: precision/layout prep
  cvt_f32_f16<<<(MTOT * D_IN / 4 + 255) / 256, 256, 0, stream>>>(
      x, xh, MTOT * D_IN / 4);
  transpose_cvt_w<<<dim3(32, 32), tb, 0, stream>>>(Wq, WqT);
  transpose_cvt_w<<<dim3(32, 32), tb, 0, stream>>>(Wk, WkT);
  transpose_cvt_w<<<dim3(32, 32), tb, 0, stream>>>(Wv, WvT);
  transpose_cvt_w<<<dim3(32, 32), tb, 0, stream>>>(Wo, WoT);

  // Stage 1: fused QKV projection (WMMA)
  qkv_gemm<<<512, 256, 0, stream>>>(xh, WqT, WkT, WvT, Qb, Kb, Vb);

  // Stage 1.5: V -> V^T per head
  transpose_v<<<dim3(2, 64, 32), tb, 0, stream>>>(Vb, VTb);

  // Stage 2: causal flash attention (WMMA + online softmax)
  flash_attn<<<512, 256, 0, stream>>>(Qb, Kb, VTb, AOb);

  // Stage 3: output projection + bias (WMMA)
  out_gemm<<<512, 256, 0, stream>>>(AOb, WoT, bo, out);
}
